// EpisodicMemory_85925115724024
// MI455X (gfx1250) — compile-verified, matrix operand order checked by reading, not run on hardware
//
#include <hip/hip_runtime.h>
#include <hip/hip_bf16.h>
#include <math.h>

// Problem constants (match reference)
#define A_ 4
#define T_ 256
#define B_ 64
#define EM_ 1024
#define ENTO_ 1024
#define IN_ 4096   // ENTO + 3*EM

// Persistent-kernel grid: 32 WGs x 8 wave32 = 256 waves = 256 16x16 output
// tiles of the per-phase [64 x 1024] GEMM (4 M-tiles x 64 N-tiles).
#define NWG 32
#define TPB 256

typedef __attribute__((ext_vector_type(16))) __bf16 v16bf;
typedef __attribute__((ext_vector_type(8)))  __bf16 v8bf;
typedef __attribute__((ext_vector_type(8)))  float  v8f;

// Global (addrspace 1) pointer types: force global_load_b128 / global_store
// instead of flat_* (flat uses LOADcnt+DScnt and the LDS path).
typedef const __attribute__((address_space(1))) v8bf*          gc_v8bf;
typedef       __attribute__((address_space(1))) float*         g_f32;
typedef       __attribute__((address_space(1))) unsigned short* g_u16;

union Frag { v16bf v; v8bf h[2]; };

__device__ __forceinline__ unsigned short f2bf(float f) {
  union { float f; unsigned u; } c; c.f = f;
  unsigned u = c.u;
  if ((u & 0x7fffffffu) > 0x7f800000u)          // NaN -> quiet bf16 NaN
    return (unsigned short)((u >> 16) | 0x0040u);
  return (unsigned short)((u + 0x7fffu + ((u >> 16) & 1u)) >> 16); // RNE
}

// ---------------------------------------------------------------------------
// fp32 -> bf16 grid-stride converter (weights, inputs, h0) — run once/launch
// ---------------------------------------------------------------------------
__global__ void cvt_f32_bf16(const float* __restrict__ in,
                             unsigned short* __restrict__ out, int n) {
  for (int i = blockIdx.x * blockDim.x + threadIdx.x; i < n;
       i += gridDim.x * blockDim.x)
    out[i] = f2bf(in[i]);
}

__global__ void init_barrier(unsigned* bar) { bar[0] = 0u; bar[1] = 0u; }

// Device-wide sense-reversing barrier (agent scope), bar[0]=count bar[1]=gen
__device__ __forceinline__ void grid_barrier(unsigned* bar, int nwg) {
  __syncthreads();
  if (threadIdx.x == 0) {
    unsigned gen = __hip_atomic_load(&bar[1], __ATOMIC_RELAXED,
                                     __HIP_MEMORY_SCOPE_AGENT);
    unsigned arrived = __hip_atomic_fetch_add(&bar[0], 1u, __ATOMIC_ACQ_REL,
                                              __HIP_MEMORY_SCOPE_AGENT);
    if (arrived == (unsigned)(nwg - 1)) {
      __hip_atomic_store(&bar[0], 0u, __ATOMIC_RELAXED,
                         __HIP_MEMORY_SCOPE_AGENT);
      __hip_atomic_fetch_add(&bar[1], 1u, __ATOMIC_RELEASE,
                             __HIP_MEMORY_SCOPE_AGENT);
    } else {
      while (__hip_atomic_load(&bar[1], __ATOMIC_ACQUIRE,
                               __HIP_MEMORY_SCOPE_AGENT) == gen)
        __builtin_amdgcn_s_sleep(1);
    }
  }
  __syncthreads();
}

// One K=1024 segment of the per-phase GEMM: 32 k-steps of
// v_wmma_f32_16x16x32_bf16 accumulating into acc.
// A fragment (16x32 bf16): lane holds row m0+lrow; lanes>=16 take K+8 / K+24
// halves (ISA §7.12.2 16-bit striping). B fragment (32x16): lane holds
// weight row (output col) n0+lrow with the same K striping.
__device__ __forceinline__ v8f mma_seg(v8f acc,
                                       const unsigned short* __restrict__ src,
                                       const unsigned short* __restrict__ w,
                                       int ldw, int m0, int n0, int lrow,
                                       int half) {
  const unsigned short* ap = src + (size_t)(m0 + lrow) * EM_ + half * 8;
  const unsigned short* bp = w + (size_t)(n0 + lrow) * (size_t)ldw + half * 8;
#pragma unroll 8
  for (int kb = 0; kb < 1024; kb += 32) {
    Frag fa, fb;
    fa.h[0] = *(gc_v8bf)(ap + kb);        // K .. K+7
    fa.h[1] = *(gc_v8bf)(ap + kb + 16);   // K+16 .. K+23
    fb.h[0] = *(gc_v8bf)(bp + kb);
    fb.h[1] = *(gc_v8bf)(bp + kb + 16);
    __builtin_prefetch(bp + kb + 512, 0, 3);  // stream weight lines ahead
    acc = __builtin_amdgcn_wmma_f32_16x16x32_bf16(
        false, fa.v, false, fb.v, (short)0, acc, false, false);
  }
  return acc;
}

// ---------------------------------------------------------------------------
// Persistent coupled-RNN kernel: 1024 serial phases (t,a), each a
// M=64, N=1024, K=5120 bf16 WMMA GEMM + tanh, separated by grid barriers.
// ---------------------------------------------------------------------------
__global__ void __launch_bounds__(TPB, 1)
rnn_persistent(const unsigned short* __restrict__ Xbf,   // [T][B][ENTO] bf16
               const unsigned short* __restrict__ Wih,   // [A][EM][IN]  bf16
               const unsigned short* __restrict__ Whh,   // [A][EM][EM]  bf16
               unsigned short* __restrict__ Hbf,         // [2][A][B][EM] bf16
               float* __restrict__ out,                  // [A][T][B][EM] f32
               unsigned* __restrict__ bar) {
  const int lane  = threadIdx.x & 31;
  const int wave  = threadIdx.x >> 5;
  const int gw    = blockIdx.x * (TPB / 32) + wave;  // 0..255
  const int mtile = gw & 3;        // 4 M-tiles of 16 rows (batch)
  const int ntile = gw >> 2;       // 64 N-tiles of 16 cols
  const int m0 = mtile * 16, n0 = ntile * 16;
  const int lrow = lane & 15;
  const int half = lane >> 4;

  for (int t = 0; t < T_; ++t) {
    for (int a = 0; a < A_; ++a) {
      // Hidden-state double buffering: h[p] has been updated (t+1) times if
      // p<a else t times; read buffer = update_count & 1.
      auto hptr = [&](int p) -> const unsigned short* {
        int buf = ((p < a) ? (t + 1) : t) & 1;
        return Hbf + ((size_t)buf * A_ + p) * (B_ * EM_);
      };
      // Concat order per reference: [x_t, h_p for p != a ascending], then the
      // W_hh * h_a term. Straight-line (no pointer arrays) so address-space
      // inference stays intact.
      const int o0 = (a == 0) ? 1 : 0;
      const int o1 = (a <= 1) ? 2 : 1;
      const int o2 = (a == 3) ? 2 : 3;
      const unsigned short* WihA = Wih + (size_t)a * (EM_ * (size_t)IN_);
      const unsigned short* WhhA = Whh + (size_t)a * (EM_ * EM_);

      v8f acc = {0.f, 0.f, 0.f, 0.f, 0.f, 0.f, 0.f, 0.f};
      acc = mma_seg(acc, Xbf + (size_t)t * (B_ * ENTO_), WihA, IN_,
                    m0, n0, lrow, half);
      acc = mma_seg(acc, hptr(o0), WihA + 1024, IN_, m0, n0, lrow, half);
      acc = mma_seg(acc, hptr(o1), WihA + 2048, IN_, m0, n0, lrow, half);
      acc = mma_seg(acc, hptr(o2), WihA + 3072, IN_, m0, n0, lrow, half);
      acc = mma_seg(acc, hptr(a), WhhA, EM_, m0, n0, lrow, half);

      // Epilogue: tanh, write fp32 output + bf16 next-state (buffer (t+1)&1).
      // C/D layout: lane 0-15 -> N=lane, M=r; lane 16-31 -> N=lane-16, M=8+r.
      unsigned short* hnew =
          Hbf + ((size_t)((t + 1) & 1) * A_ + a) * (B_ * EM_);
      float* outp = out + (((size_t)a * T_ + t) * B_) * EM_;
      const int n = n0 + lrow;
#pragma unroll
      for (int r = 0; r < 8; ++r) {
        int m = m0 + half * 8 + r;
        float v = tanhf(acc[r]);
        *(g_f32)(outp + (size_t)m * EM_ + n) = v;
        *(g_u16)(hnew + (size_t)m * EM_ + n) = f2bf(v);
      }

      grid_barrier(bar, NWG);  // phase (t,a) fully visible before next phase
    }
  }
}

// ---------------------------------------------------------------------------
// Launch. Inputs: d_in[0]=ento_feat [T,B,ENTO] f32, d_in[1]=h0 [A,B,EM] f32,
// d_in[2]=W_ih [A,EM,IN] f32, d_in[3]=W_hh [A,EM,EM] f32.
// Output: [A,T,B,EM] f32.  Workspace: ~77 MB bf16 staging + barrier words.
// ---------------------------------------------------------------------------
extern "C" void kernel_launch(void* const* d_in, const int* in_sizes, int n_in,
                              void* d_out, int out_size, void* d_ws,
                              size_t ws_size, hipStream_t stream) {
  const float* x_f32   = (const float*)d_in[0];
  const float* h0_f32  = (const float*)d_in[1];
  const float* wih_f32 = (const float*)d_in[2];
  const float* whh_f32 = (const float*)d_in[3];

  char* ws = (char*)d_ws;
  const size_t oBar = 0;
  const size_t oWih = 256;
  const size_t szWih = (size_t)A_ * EM_ * IN_ * 2;      // 32 MB
  const size_t oWhh = oWih + szWih;
  const size_t szWhh = (size_t)A_ * EM_ * EM_ * 2;      // 8 MB
  const size_t oX = oWhh + szWhh;
  const size_t szX = (size_t)T_ * B_ * ENTO_ * 2;       // 32 MB
  const size_t oH = oX + szX;                           // 2 x [A,B,EM] bf16

  unsigned*       bar = (unsigned*)(ws + oBar);
  unsigned short* Wih = (unsigned short*)(ws + oWih);
  unsigned short* Whh = (unsigned short*)(ws + oWhh);
  unsigned short* Xbf = (unsigned short*)(ws + oX);
  unsigned short* Hbf = (unsigned short*)(ws + oH);

  init_barrier<<<1, 1, 0, stream>>>(bar);

  const int cvtBlocks = 2048;
  cvt_f32_bf16<<<cvtBlocks, 256, 0, stream>>>(wih_f32, Wih, A_ * EM_ * IN_);
  cvt_f32_bf16<<<cvtBlocks, 256, 0, stream>>>(whh_f32, Whh, A_ * EM_ * EM_);
  cvt_f32_bf16<<<cvtBlocks, 256, 0, stream>>>(x_f32, Xbf, T_ * B_ * ENTO_);
  // h0 -> buffer 0 (parity logic reads buf (update_count & 1), starts at 0)
  cvt_f32_bf16<<<cvtBlocks, 256, 0, stream>>>(h0_f32, Hbf, A_ * B_ * EM_);

  rnn_persistent<<<NWG, TPB, 0, stream>>>(Xbf, Wih, Whh, Hbf, (float*)d_out,
                                          bar);
  (void)in_sizes; (void)n_in; (void)out_size; (void)ws_size;
}
